// Head_23622320128652
// MI455X (gfx1250) — matrix operand order repaired and checked by reading, not compiled, not run on hardware
//
#include <hip/hip_runtime.h>
#include <hip/hip_bf16.h>

// ---------------------------------------------------------------------------
// Causal single-head attention, CDNA5 (gfx1250), wave32, bf16 WMMA path.
//   B=512, T=256, C=384, H=64
// Kernel 1: q = (x@wq)*C^-0.5, k = x@wk, v^T = (x@wv)^T  (bf16 -> d_ws)
//           x tile + weights staged in LDS (padded pitches), WMMA bf16,
//           epilogue repacked through LDS for coalesced b128 stores.
// Kernel 2: flash-style causal softmax(q k^T) v via S^T = K Q^T trick;
//           K/V^T staged via async global->LDS (ASYNCcnt) when available.
// ---------------------------------------------------------------------------

#define NB 512
#define NT 256
#define NC 384
#define NH 64

#define WP  392   // LDS pitch (elems) for weight rows (NC=384 + 8 pad)
#define XP  392   // LDS pitch (elems) for x-tile rows
#define KP  72    // LDS pitch (elems) for K rows   (NH=64 + 8 pad)
#define VP  264   // LDS pitch (elems) for V^T rows (NT=256 + 8 pad)
#define FP  68    // LDS pitch (floats) for out-staging rows (NH=64 + 4 pad)

typedef __attribute__((ext_vector_type(16))) __bf16         v16bf;
typedef __attribute__((ext_vector_type(8)))  __bf16         v8bf;
typedef __attribute__((ext_vector_type(4)))  __bf16         v4bf;
typedef __attribute__((ext_vector_type(16))) float          v16f;
typedef __attribute__((ext_vector_type(8)))  float          v8f;
typedef __attribute__((ext_vector_type(4)))  float          v4f;
typedef __attribute__((ext_vector_type(16))) unsigned short v16us;
typedef __attribute__((ext_vector_type(8)))  unsigned short v8us;
typedef __attribute__((ext_vector_type(4)))  unsigned int   v4u;
typedef __attribute__((ext_vector_type(2)))  unsigned int   v2u;
typedef __attribute__((ext_vector_type(4)))  int            v4i;

// pointers to address-space-qualified v4i (for the async-to-LDS builtin)
typedef __attribute__((address_space(1))) v4i* gv4p;   // global (AS1)
typedef __attribute__((address_space(3))) v4i* lv4p;   // LDS (AS3)

union ABfrag { v16bf bf; v16us us; v4u q[2]; };
union O8     { v8bf bf; v8us us; };
union C4     { v4bf bf; v2u u2; };

__device__ __forceinline__ unsigned short f2bf(float f) {
    return __builtin_bit_cast(unsigned short, (__bf16)f);  // v_cvt on gfx1250
}

__device__ __forceinline__ v8f zero8() {
    v8f z = {0.f, 0.f, 0.f, 0.f, 0.f, 0.f, 0.f, 0.f};
    return z;
}

__device__ __forceinline__ v8f bf16_wmma(const ABfrag& a, const ABfrag& b, v8f c) {
    return __builtin_amdgcn_wmma_f32_16x16x32_bf16(false, a.bf, false, b.bf,
                                                   (short)0, c, false, false);
}

#if __has_builtin(__builtin_amdgcn_global_load_async_to_lds_b128) && \
    __has_builtin(__builtin_amdgcn_s_wait_asynccnt)
#define USE_ASYNC_LDS 1
__device__ __forceinline__ void async_copy16(const void* g, const void* l) {
    // generic LDS pointers carry the AS3 offset in their low 32 bits
    __builtin_amdgcn_global_load_async_to_lds_b128(
        (gv4p)(unsigned long long)g,
        (lv4p)(unsigned int)(unsigned long long)l,
        0, 0);
}
#else
#define USE_ASYNC_LDS 0
#endif

// ---------------------------------------------------------------------------
// Kernel 1: QKV projection. 256 threads = 8 waves, 128 token rows per block.
// LDS: wT[3][NH][WP] bf16 (150528 B) + x[128][XP] bf16 (100352 B) = 250880 B.
// Each wave: 16 rows, 12 K-steps x 12 WMMAs (q,k,v x 4 h-tiles, shared A).
// Epilogue aliases the wave's own (dead) x-tile rows as repack scratch.
// ---------------------------------------------------------------------------
__global__ void __launch_bounds__(256) proj_qkv_kernel(
    const float* __restrict__ x,
    const float* __restrict__ wq,
    const float* __restrict__ wk,
    const float* __restrict__ wv,
    unsigned short* __restrict__ qo,   // [NB*NT][NH] bf16 (scale folded in)
    unsigned short* __restrict__ ko,   // [NB*NT][NH] bf16
    unsigned short* __restrict__ vto)  // [NB][NH][NT] bf16 (transposed)
{
    extern __shared__ __align__(16) unsigned short lds[];
    const int XBASE = 3 * NH * WP;                    // 75264 elems
    const int tid = threadIdx.x;

    // ---- stage weights: fp32 (C,H) row-major -> bf16 (H,C) transposed ----
    {
        const float* wsrc[3] = {wq, wk, wv};
        #pragma unroll
        for (int m = 0; m < 3; ++m) {
            const float* w = wsrc[m];
            unsigned short* dst = lds + m * (NH * WP);
            for (int i = tid; i < NC * NH; i += 256) {
                const int c = i >> 6;                 // / NH
                const int h = i & (NH - 1);
                dst[h * WP + c] = f2bf(w[i]);
            }
        }
    }
    // ---- stage x tile: 128 rows, coalesced float4 reads, bf16 in LDS ----
    {
        const float4* xs = (const float4*)(x + (size_t)blockIdx.x * 128 * NC);
        v2u* xd = (v2u*)(lds + XBASE);
        for (int i = tid; i < 128 * (NC / 4); i += 256) {
            const int row = i / (NC / 4);
            const int c4  = i - row * (NC / 4);
            const float4 f = xs[i];
            v4f vf = {f.x, f.y, f.z, f.w};
            C4 cc;
            cc.bf = __builtin_convertvector(vf, v4bf);
            xd[row * (XP / 4) + c4] = cc.u2;
        }
    }
    __syncthreads();

    const int lane = tid & 31;
    const int wave = tid >> 5;
    const int lr   = lane & 15;
    const int hiK  = lane >> 4;

    v8f acc[3][4];
    #pragma unroll
    for (int m = 0; m < 3; ++m)
        #pragma unroll
        for (int ht = 0; ht < 4; ++ht) acc[m][ht] = zero8();

    const unsigned short* xrow = lds + XBASE + (size_t)(wave * 16 + lr) * XP;

    for (int ki = 0; ki < NC / 32; ++ki) {
        // A fragment (16x32): lane half 0 -> K 0..15, half 1 -> K 16..31
        ABfrag a;
        const int ca = ki * 32 + hiK * 16;
        a.q[0] = *(const v4u*)(xrow + ca);
        a.q[1] = *(const v4u*)(xrow + ca + 8);
        // B fragments (32x16): lane half 0 -> K 0..15, half 1 -> K 16..31
        const int cb = ki * 32 + hiK * 16;
        #pragma unroll
        for (int m = 0; m < 3; ++m) {
            #pragma unroll
            for (int ht = 0; ht < 4; ++ht) {
                ABfrag bfr;
                const unsigned short* p = lds + m * (NH * WP) + (ht * 16 + lr) * WP + cb;
                bfr.q[0] = *(const v4u*)p;
                bfr.q[1] = *(const v4u*)(p + 8);
                acc[m][ht] = bf16_wmma(a, bfr, acc[m][ht]);
            }
        }
    }

    // ---- epilogue: repack via own x-region, then coalesced b128 stores ----
    const float qscale = 0.05103103630798288f;        // 384^-0.5 folded into q
    const int row0 = blockIdx.x * 128 + wave * 16;
    const int bb   = row0 >> 8;                        // batch
    const int t0c  = row0 & (NT - 1);
    unsigned short* eq = lds + XBASE + wave * (16 * XP); // wave-private scratch
    unsigned short* ek = eq + 1024;
    unsigned short* ev = ek + 1024;

    #pragma unroll
    for (int ht = 0; ht < 4; ++ht) {
        O8 qb, kb, vb;
        qb.bf = __builtin_convertvector(acc[0][ht] * qscale, v8bf);
        kb.bf = __builtin_convertvector(acc[1][ht], v8bf);
        vb.bf = __builtin_convertvector(acc[2][ht], v8bf);
        #pragma unroll
        for (int r = 0; r < 8; ++r) {
            const int tl = r + hiK * 8;                // C layout: M = r + 8*half
            const int h  = ht * 16 + lr;               // N = lane%16
            eq[tl * 64 + h] = qb.us[r];                // [t][h]
            ek[tl * 64 + h] = kb.us[r];                // [t][h]
            ev[h * 16 + tl] = vb.us[r];                // [h][t]
        }
    }
    // same-wave LDS is in order; wave-private region, no barrier needed
    {
        const v4u* s = (const v4u*)eq;
        v4u* d = (v4u*)(qo + (size_t)row0 * NH);
        #pragma unroll
        for (int i = 0; i < 4; ++i) d[i * 32 + lane] = s[i * 32 + lane];
    }
    {
        const v4u* s = (const v4u*)ek;
        v4u* d = (v4u*)(ko + (size_t)row0 * NH);
        #pragma unroll
        for (int i = 0; i < 4; ++i) d[i * 32 + lane] = s[i * 32 + lane];
    }
    {
        const v4u* s = (const v4u*)ev;
        #pragma unroll
        for (int i = 0; i < 4; ++i) {
            const int c = i * 32 + lane;
            const int h = c >> 1, half = c & 1;
            *(v4u*)(vto + ((size_t)(bb * NH + h)) * NT + t0c + half * 8) = s[c];
        }
    }
}

// ---------------------------------------------------------------------------
// Kernel 2: causal attention. One workgroup = (batch, 64 t-rows); 4 waves of
// 16 rows. K[NT][KP] and V^T[NH][VP] staged in LDS (async path if available).
// Per 32-wide s-block: S^T = K Q^T (4 WMMAs), mask, online softmax (xor-16),
// O^T += V^T P^T (4 WMMAs). Out repacked via LDS for coalesced b128 stores.
// ---------------------------------------------------------------------------
__global__ void __launch_bounds__(128) attn_kernel(
    const unsigned short* __restrict__ q,    // [NB*NT][NH] bf16 (pre-scaled)
    const unsigned short* __restrict__ k,    // [NB*NT][NH] bf16
    const unsigned short* __restrict__ vt,   // [NB][NH][NT] bf16
    float* __restrict__ out)                 // [NB][NT][NH] f32
{
    extern __shared__ __align__(16) unsigned short smem[];
    unsigned short* kl = smem;                 // [NT][KP]
    unsigned short* vl = smem + NT * KP;       // [NH][VP]
    float* fstage = (float*)(smem + NT * KP + NH * VP);  // [4 waves][16][FP]

    const int b    = blockIdx.x >> 2;
    const int tb0  = (blockIdx.x & 3) * 64;
    const int tid  = threadIdx.x;
    const int smax = tb0 + 64;                 // causal: only s < smax touched

    {   // stage K rows [0,smax) and V^T columns [0,smax), padded pitches
        const v4u* kg = (const v4u*)(k + (size_t)b * NT * NH);
        const v4u* vg = (const v4u*)(vt + (size_t)b * NH * NT);
        const int rowq = smax >> 3;            // 16B chunks per V^T row
#if USE_ASYNC_LDS
        for (int i = tid; i < smax * 8; i += 128) {
            const int s = i >> 3, j = i & 7;
            async_copy16(kg + i, kl + s * KP + j * 8);
        }
        for (int i = tid; i < NH * rowq; i += 128) {
            const int h = i / rowq, j = i - h * rowq;
            async_copy16(vg + h * (NT / 8) + j, vl + h * VP + j * 8);
        }
        __builtin_amdgcn_s_wait_asynccnt(0);
#else
        for (int i = tid; i < smax * 8; i += 128) {
            const int s = i >> 3, j = i & 7;
            *(v4u*)(kl + s * KP + j * 8) = kg[i];
        }
        for (int i = tid; i < NH * rowq; i += 128) {
            const int h = i / rowq, j = i - h * rowq;
            *(v4u*)(vl + h * VP + j * 8) = vg[h * (NT / 8) + j];
        }
#endif
    }
    __syncthreads();

    const int lane  = tid & 31;
    const int wave  = tid >> 5;
    const int lr    = lane & 15;
    const int hiK   = lane >> 4;
    const int ttile = tb0 + wave * 16;
    const int tg    = ttile + lr;              // this lane's t column

    // Q^T B-fragments (h x t): lane half 0 -> K 0..15, half 1 -> 16..31
    ABfrag qf[2];
    {
        const unsigned short* qrow = q + ((size_t)b * NT + tg) * NH;
        #pragma unroll
        for (int hk = 0; hk < 2; ++hk) {
            const int hb = hk * 32 + hiK * 16;
            qf[hk].q[0] = *(const v4u*)(qrow + hb);
            qf[hk].q[1] = *(const v4u*)(qrow + hb + 8);
        }
    }

    v8f O[4];
    #pragma unroll
    for (int hh = 0; hh < 4; ++hh) O[hh] = zero8();
    float Mx = -INFINITY, Sm = 0.f;

    const int nblk = (ttile + 47) >> 5;        // s-blocks of 32, sb <= ttile
    for (int bi = 0; bi < nblk; ++bi) {
        const int sb = bi * 32;

        // S^T tiles (M = s_local, N = t)
        v8f st0 = zero8(), st1 = zero8();
        #pragma unroll
        for (int hk = 0; hk < 2; ++hk) {
            const unsigned short* kr = kl + (size_t)(sb + lr) * KP + hk * 32 + hiK * 16;
            ABfrag kf0, kf1;
            kf0.q[0] = *(const v4u*)kr;
            kf0.q[1] = *(const v4u*)(kr + 8);
            kf1.q[0] = *(const v4u*)(kr + 16 * KP);
            kf1.q[1] = *(const v4u*)(kr + 16 * KP + 8);
            st0 = bf16_wmma(kf0, qf[hk], st0);
            st1 = bf16_wmma(kf1, qf[hk], st1);
        }

        if (sb + 31 > ttile) {                 // causal mask, diagonal blocks only
            const int srow = sb + hiK * 8;
            #pragma unroll
            for (int r = 0; r < 8; ++r) {
                if (srow + r      > tg) st0[r] = -INFINITY;
                if (srow + 16 + r > tg) st1[r] = -INFINITY;
            }
        }

        // online softmax: row t lives on lane pair {t, t+16}
        float bm = -INFINITY;
        #pragma unroll
        for (int r = 0; r < 8; ++r) bm = fmaxf(bm, fmaxf(st0[r], st1[r]));
        bm = fmaxf(bm, __shfl_xor(bm, 16));
        const float nM    = fmaxf(Mx, bm);
        const float alpha = __expf(Mx - nM);

        float p0[8], p1[8], ssum = 0.f;
        #pragma unroll
        for (int r = 0; r < 8; ++r) {
            p0[r] = __expf(st0[r] - nM);
            p1[r] = __expf(st1[r] - nM);
            ssum += p0[r] + p1[r];
        }
        ssum += __shfl_xor(ssum, 16);
        Sm = Sm * alpha + ssum;
        Mx = nM;
        #pragma unroll
        for (int hh = 0; hh < 4; ++hh) O[hh] *= alpha;

        // P^T B-fragment: half 0 needs s_local 0..15, half 1 needs 16..31;
        // own C-layout rows + partner's via one xor-16 exchange, then pk-cvt.
        v16f pv;
        #pragma unroll
        for (int r = 0; r < 8; ++r) {
            const float o0 = __shfl_xor(p0[r], 16);
            const float o1 = __shfl_xor(p1[r], 16);
            pv[r]     = hiK ? o1    : p0[r];
            pv[8 + r] = hiK ? p1[r] : o0;
        }
        ABfrag pf;
        pf.bf = __builtin_convertvector(pv, v16bf);

        // O^T += V^T P^T  (A = V^T rows h, contiguous-K A layout from LDS)
        #pragma unroll
        for (int hh = 0; hh < 4; ++hh) {
            ABfrag vf;
            const unsigned short* vr = vl + (size_t)(hh * 16 + lr) * VP + sb + hiK * 16;
            vf.q[0] = *(const v4u*)vr;
            vf.q[1] = *(const v4u*)(vr + 8);
            O[hh] = bf16_wmma(vf, pf, O[hh]);
        }
    }

    // ---- epilogue: O^T (M=h, N=t) -> LDS [t][h] (padded) -> coalesced out ----
    const float inv = 1.0f / Sm;
    float* fo = fstage + wave * (16 * FP);     // wave-private
    #pragma unroll
    for (int hh = 0; hh < 4; ++hh)
        #pragma unroll
        for (int r = 0; r < 8; ++r)
            fo[lr * FP + hh * 16 + r + hiK * 8] = O[hh][r] * inv;

    float* od = out + ((size_t)b * NT + ttile) * NH;   // 16 x 64 contiguous
    #pragma unroll
    for (int i = 0; i < 8; ++i) {
        const int c = i * 32 + lane;           // float4 chunk 0..255
        const int t = c >> 4, j = c & 15;
        const v4f val = *(const v4f*)(fo + t * FP + j * 4);
        *(v4f*)(od + c * 4) = val;
    }
}

// ---------------------------------------------------------------------------
extern "C" void kernel_launch(void* const* d_in, const int* in_sizes, int n_in,
                              void* d_out, int out_size, void* d_ws, size_t ws_size,
                              hipStream_t stream) {
    const float* x  = (const float*)d_in[0];
    const float* wq = (const float*)d_in[1];
    const float* wk = (const float*)d_in[2];
    const float* wv = (const float*)d_in[3];
    float* out = (float*)d_out;

    // workspace: q, k (row-major) and v^T, all bf16
    unsigned short* qws = (unsigned short*)d_ws;
    unsigned short* kws = qws + (size_t)NB * NT * NH;
    unsigned short* vws = kws + (size_t)NB * NT * NH;

    const int proj_blocks = (NB * NT) / 128;                       // 1024
    const size_t proj_lds = (size_t)(3 * NH * WP + 128 * XP) * 2;  // 250880 B
    proj_qkv_kernel<<<proj_blocks, 256, proj_lds, stream>>>(x, wq, wk, wv,
                                                            qws, kws, vws);

    const int attn_blocks = NB * (NT / 64);                        // 2048
    const size_t attn_lds = (size_t)(NT * KP + NH * VP) * 2        // 70656 B
                          + (size_t)4 * 16 * FP * 4;               // +17408 B
    attn_kernel<<<attn_blocks, 128, attn_lds, stream>>>(qws, kws, vws, out);
}
